// MultiHeadAttention_89429809037632
// MI455X (gfx1250) — matrix operand order repaired and checked by reading, compiled, not run on hardware
//
#include <hip/hip_runtime.h>

#define NB 4
#define LSEQ 4096
#define DMODEL 2048
#define NH 16
#define HP 128
#define LDSW 40   // padded LDS row stride in bf16 (80 bytes)

typedef __attribute__((ext_vector_type(16))) __bf16 bf16x16;
typedef __attribute__((ext_vector_type(8)))  __bf16 bf16x8;
typedef __attribute__((ext_vector_type(4)))  __bf16 bf16x4;
typedef __attribute__((ext_vector_type(8)))  float  f32x8;
typedef __attribute__((ext_vector_type(4)))  float  f32x4;

__device__ __forceinline__ float softplusf(float x) {
  // log(1+exp(x)); for x>15 the correction is < 3e-7 relative
  return (x > 15.0f) ? x : log1pf(__expf(x));
}

__device__ __forceinline__ bf16x16 combine16(bf16x8 lo, bf16x8 hi) {
  bf16x16 r;
#pragma unroll
  for (int i = 0; i < 8; ++i) { r[i] = lo[i]; r[8 + i] = hi[i]; }
  return r;
}

// ---------------------------------------------------------------------------
// Kernel A: y = x @ W^T  (M = NB*LSEQ rows, N = DMODEL cols, K = DMODEL)
// mode 0: y=softplus(y) -> pq   row-major (N,Lq,D), bf16
// mode 1: y=softplus(y), zero masked keys -> pkT  (N,H,P,Lk), bf16
// mode 2: y -> vT (N,H,P,Lk), bf16
// Block: 128x128 tile, 256 threads (8 waves as 4x2), wave tile 32x64,
// K-step 32, LDS-staged bf16, next-chunk global prefetch.
// ---------------------------------------------------------------------------
__global__ __launch_bounds__(256)
void proj_kernel(const float* __restrict__ X, const float* __restrict__ Wm,
                 __bf16* __restrict__ outN, __bf16* __restrict__ outT,
                 const unsigned char* __restrict__ kmask, int mode) {
  __shared__ __align__(16) __bf16 lA[128 * LDSW];
  __shared__ __align__(16) __bf16 lB[128 * LDSW];

  const int tid   = threadIdx.x;
  const int wave  = tid >> 5;
  const int lane  = tid & 31;
  const int lhalf = lane >> 4;
  const int l16   = lane & 15;
  const int wm    = wave >> 1;   // 0..3 : rows [wm*32, +32)
  const int wn    = wave & 1;    // 0..1 : cols [wn*64, +64)

  const long rowBase = (long)blockIdx.x * 128;   // over NB*LSEQ
  const long colBase = (long)blockIdx.y * 128;   // over DMODEL

  f32x8 acc[2][4] = {};

  const int ldRow  = tid >> 1;   // 0..127
  const int ldHalf = tid & 1;    // 16-col half
  const float* srcX = X  + (rowBase + ldRow) * (long)DMODEL + ldHalf * 16;
  const float* srcW = Wm + (colBase + ldRow) * (long)DMODEL + ldHalf * 16;

  for (int kc = 0; kc < DMODEL / 32; ++kc) {
    const int k0 = kc * 32;
    // stage X and W tiles (128x32 f32 -> bf16)
    {
      __bf16* dst = &lA[ldRow * LDSW + ldHalf * 16];
#pragma unroll
      for (int j = 0; j < 4; ++j) {
        f32x4 v = *(const f32x4*)(srcX + k0 + j * 4);
        bf16x4 b;
#pragma unroll
        for (int e = 0; e < 4; ++e) b[e] = (__bf16)v[e];
        *(bf16x4*)(dst + j * 4) = b;
      }
    }
    {
      __bf16* dst = &lB[ldRow * LDSW + ldHalf * 16];
#pragma unroll
      for (int j = 0; j < 4; ++j) {
        f32x4 v = *(const f32x4*)(srcW + k0 + j * 4);
        bf16x4 b;
#pragma unroll
        for (int e = 0; e < 4; ++e) b[e] = (__bf16)v[e];
        *(bf16x4*)(dst + j * 4) = b;
      }
    }
    if (kc + 1 < DMODEL / 32) {       // hide next chunk's HBM latency
      __builtin_prefetch(srcX + k0 + 32, 0, 3);
      __builtin_prefetch(srcW + k0 + 32, 0, 3);
    }
    __syncthreads();

    // load ALL fragments first -> one ds wait, 12 b128 loads per 8 WMMAs
    bf16x16 afrag[2];
#pragma unroll
    for (int mf = 0; mf < 2; ++mf) {
      const __bf16* ap = &lA[(wm * 32 + mf * 16 + l16) * LDSW + lhalf * 8];
      afrag[mf] = combine16(*(const bf16x8*)ap, *(const bf16x8*)(ap + 16));
    }
    bf16x16 bfrag[4];
#pragma unroll
    for (int nf = 0; nf < 4; ++nf) {
      const __bf16* bp = &lB[(wn * 64 + nf * 16 + l16) * LDSW + lhalf * 8];
      bfrag[nf] = combine16(*(const bf16x8*)bp, *(const bf16x8*)(bp + 16));
    }
#pragma unroll
    for (int nf = 0; nf < 4; ++nf)
#pragma unroll
      for (int mf = 0; mf < 2; ++mf)
        acc[mf][nf] = __builtin_amdgcn_wmma_f32_16x16x32_bf16(
            false, afrag[mf], false, bfrag[nf], (short)0, acc[mf][nf], false, false);
    __syncthreads();
  }

  if (mode == 0) {
#pragma unroll
    for (int mf = 0; mf < 2; ++mf)
#pragma unroll
      for (int nf = 0; nf < 4; ++nf) {
        const long col = colBase + wn * 64 + nf * 16 + l16;
#pragma unroll
        for (int v = 0; v < 8; ++v) {
          const long rm = rowBase + wm * 32 + mf * 16 + lhalf * 8 + v;
          outN[rm * (long)DMODEL + col] = (__bf16)softplusf(acc[mf][nf][v]);
        }
      }
  } else {
    // transposed store: lane owns fixed col, 8 consecutive L-rows -> one b128
#pragma unroll
    for (int mf = 0; mf < 2; ++mf) {
      const long rm0 = rowBase + wm * 32 + mf * 16 + lhalf * 8;
      const int nb  = (int)(rm0 / LSEQ);
      const int lk0 = (int)(rm0 % LSEQ);
#pragma unroll
      for (int nf = 0; nf < 4; ++nf) {
        const int col = (int)colBase + wn * 64 + nf * 16 + l16;
        const int h = col >> 7, p = col & 127;
        bf16x8 ov;
#pragma unroll
        for (int v = 0; v < 8; ++v) {
          float val = acc[mf][nf][v];
          if (mode == 1) {
            val = softplusf(val);
            if (kmask[nb * LSEQ + lk0 + v]) val = 0.0f;
          }
          ov[v] = (__bf16)val;
        }
        *(bf16x8*)&outT[(((long)nb * NH + h) * HP + p) * (long)LSEQ + lk0] = ov;
      }
    }
  }
}

// ---------------------------------------------------------------------------
// Kernel B: per (n,h): kv[d,e] = sum_lk pk[lk,d]*v[lk,e] ; ksum[d] = sum pk[lk,d]
// Inputs pkT/vT are lk-contiguous -> both WMMA operands stream contiguously.
// Output stored transposed: kvT[e,d] (d contiguous) for the next GEMM's B-frags.
// ---------------------------------------------------------------------------
__global__ __launch_bounds__(256)
void kv_kernel(const __bf16* __restrict__ pkT, const __bf16* __restrict__ vT,
               __bf16* __restrict__ kvT, float* __restrict__ ksum) {
  __shared__ __align__(16) __bf16 lP[128 * LDSW];
  __shared__ __align__(16) __bf16 lV[128 * LDSW];

  const int tid = threadIdx.x;
  const int wave = tid >> 5, lane = tid & 31, lhalf = lane >> 4, l16 = lane & 15;
  const int wm = wave >> 1;   // d rows [wm*32, +32)
  const int wn = wave & 1;    // e cols [wn*64, +64)
  const int nh = blockIdx.x;
  const long headP = (long)nh * HP * LSEQ;

  f32x8 acc[2][4] = {};
  float srow[2] = {0.0f, 0.0f};

  const int ldRow = tid >> 1, ldHalf = tid & 1;
  const __bf16* srcP = pkT + headP + (long)ldRow * LSEQ + ldHalf * 16;
  const __bf16* srcV = vT  + headP + (long)ldRow * LSEQ + ldHalf * 16;

  for (int kc = 0; kc < LSEQ / 32; ++kc) {
    const int k0 = kc * 32;
    {
      __bf16* dst = &lP[ldRow * LDSW + ldHalf * 16];
      *(bf16x8*)dst       = *(const bf16x8*)(srcP + k0);
      *(bf16x8*)(dst + 8) = *(const bf16x8*)(srcP + k0 + 8);
    }
    {
      __bf16* dst = &lV[ldRow * LDSW + ldHalf * 16];
      *(bf16x8*)dst       = *(const bf16x8*)(srcV + k0);
      *(bf16x8*)(dst + 8) = *(const bf16x8*)(srcV + k0 + 8);
    }
    if (kc + 1 < LSEQ / 32) {
      __builtin_prefetch(srcP + k0 + 32, 0, 3);
      __builtin_prefetch(srcV + k0 + 32, 0, 3);
    }
    __syncthreads();

    bf16x16 afrag[2];
#pragma unroll
    for (int mf = 0; mf < 2; ++mf) {
      const __bf16* ap = &lP[(wm * 32 + mf * 16 + l16) * LDSW + lhalf * 8];
      afrag[mf] = combine16(*(const bf16x8*)ap, *(const bf16x8*)(ap + 16));
#pragma unroll
      for (int i = 0; i < 16; ++i) srow[mf] += (float)afrag[mf][i];  // ksum partial
    }
    bf16x16 bfrag[4];
#pragma unroll
    for (int nf = 0; nf < 4; ++nf) {
      const __bf16* bp = &lV[(wn * 64 + nf * 16 + l16) * LDSW + lhalf * 8];
      bfrag[nf] = combine16(*(const bf16x8*)bp, *(const bf16x8*)(bp + 16));
    }
#pragma unroll
    for (int nf = 0; nf < 4; ++nf)
#pragma unroll
      for (int mf = 0; mf < 2; ++mf)
        acc[mf][nf] = __builtin_amdgcn_wmma_f32_16x16x32_bf16(
            false, afrag[mf], false, bfrag[nf], (short)0, acc[mf][nf], false, false);
    __syncthreads();
  }

  // ksum: lanes L and L^16 hold the two K-halves of row d = wm*32 + mf*16 + l16
#pragma unroll
  for (int mf = 0; mf < 2; ++mf) {
    float tot = srow[mf] + __shfl_xor(srow[mf], 16, 32);
    if (wn == 0 && lane < 16)
      ksum[nh * HP + wm * 32 + mf * 16 + lane] = tot;
  }

  // transposed store kvT[e, d0..d0+7]
#pragma unroll
  for (int mf = 0; mf < 2; ++mf) {
    const int d0 = wm * 32 + mf * 16 + lhalf * 8;
#pragma unroll
    for (int nf = 0; nf < 4; ++nf) {
      const int e = wn * 64 + nf * 16 + l16;
      bf16x8 ov;
#pragma unroll
      for (int v = 0; v < 8; ++v) ov[v] = (__bf16)acc[mf][nf][v];
      *(bf16x8*)&kvT[((long)nh * HP + e) * HP + d0] = ov;
    }
  }
}

// ---------------------------------------------------------------------------
// Kernel C: num = pq @ kv, den = pq @ ksum, out = num/den (masked queries -> 0)
// ---------------------------------------------------------------------------
__global__ __launch_bounds__(256)
void attn_out_kernel(const __bf16* __restrict__ pq, const __bf16* __restrict__ kvT,
                     const float* __restrict__ ksum,
                     const unsigned char* __restrict__ qmask,
                     float* __restrict__ out) {
  __shared__ float lks[HP];

  const int tid = threadIdx.x;
  const int wave = tid >> 5, lane = tid & 31, lhalf = lane >> 4, l16 = lane & 15;
  const int wm = wave >> 1;   // q rows [wm*32, +32)
  const int wn = wave & 1;    // e cols [wn*64, +64)
  const int bq = blockIdx.x;            // q-tile of 128
  const int nh = blockIdx.y;            // 0..63
  const int nb = nh >> 4, h = nh & 15;

  if (tid < HP) lks[tid] = ksum[nh * HP + tid];
  __syncthreads();

  f32x8 acc[2][4] = {};
  float dpart[2] = {0.0f, 0.0f};

  for (int kc = 0; kc < HP / 32; ++kc) {
    const int k0 = kc * 32;
    bf16x16 afrag[2];
#pragma unroll
    for (int mf = 0; mf < 2; ++mf) {
      const long qrow = (long)nb * LSEQ + bq * 128 + wm * 32 + mf * 16 + l16;
      const __bf16* ap = pq + qrow * (long)DMODEL + h * HP + k0 + lhalf * 8;
      afrag[mf] = combine16(*(const bf16x8*)ap, *(const bf16x8*)(ap + 16));
      // den partial for q-row = l16 over this lane's d-subset
#pragma unroll
      for (int i = 0; i < 8; ++i) dpart[mf] += (float)afrag[mf][i]     * lks[k0 + lhalf * 8 + i];
#pragma unroll
      for (int i = 0; i < 8; ++i) dpart[mf] += (float)afrag[mf][8 + i] * lks[k0 + lhalf * 8 + 16 + i];
    }
    bf16x16 bfrag[4];
#pragma unroll
    for (int nf = 0; nf < 4; ++nf) {
      const __bf16* bp = kvT + ((long)nh * HP + wn * 64 + nf * 16 + l16) * HP + k0 + lhalf * 8;
      bfrag[nf] = combine16(*(const bf16x8*)bp, *(const bf16x8*)(bp + 16));
    }
#pragma unroll
    for (int nf = 0; nf < 4; ++nf)
#pragma unroll
      for (int mf = 0; mf < 2; ++mf)
        acc[mf][nf] = __builtin_amdgcn_wmma_f32_16x16x32_bf16(
            false, afrag[mf], false, bfrag[nf], (short)0, acc[mf][nf], false, false);
  }

#pragma unroll
  for (int mf = 0; mf < 2; ++mf)
#pragma unroll
    for (int nf = 0; nf < 4; ++nf) {
      const int e = wn * 64 + nf * 16 + l16;
#pragma unroll
      for (int v = 0; v < 8; ++v) {
        const int ml = v + 8 * lhalf;  // D-fragment row 0..15
        const float den = __shfl(dpart[mf], ml, 32) + __shfl(dpart[mf], ml + 16, 32);
        const int ql = bq * 128 + wm * 32 + mf * 16 + ml;
        float val = acc[mf][nf][v] / den;
        if (qmask[nb * LSEQ + ql]) val = 0.0f;
        out[((long)nb * LSEQ + ql) * (long)DMODEL + h * HP + e] = val;
      }
    }
}

// ---------------------------------------------------------------------------
extern "C" void kernel_launch(void* const* d_in, const int* in_sizes, int n_in,
                              void* d_out, int out_size, void* d_ws, size_t ws_size,
                              hipStream_t stream) {
  const float* query = (const float*)d_in[0];
  const float* key   = (const float*)d_in[1];
  const float* Wq    = (const float*)d_in[2];
  const float* Wk    = (const float*)d_in[3];
  const float* Wv    = (const float*)d_in[4];
  const unsigned char* qmask = (const unsigned char*)d_in[5];
  const unsigned char* kmask = (const unsigned char*)d_in[6];
  float* out = (float*)d_out;

  char* ws = (char*)d_ws;
  const size_t szProj = (size_t)NB * LSEQ * DMODEL * sizeof(__bf16);   // 67 MB each
  __bf16* pq   = (__bf16*)(ws);
  __bf16* pkT  = (__bf16*)(ws + szProj);
  __bf16* vT   = (__bf16*)(ws + 2 * szProj);
  __bf16* kvT  = (__bf16*)(ws + 3 * szProj);
  float*  ksum = (float*)(ws + 3 * szProj + (size_t)NB * NH * HP * HP * sizeof(__bf16));

  dim3 gA((NB * LSEQ) / 128, DMODEL / 128);
  proj_kernel<<<gA, 256, 0, stream>>>(query, Wq, pq, nullptr, nullptr, 0);
  proj_kernel<<<gA, 256, 0, stream>>>(key,   Wk, nullptr, pkT, kmask, 1);
  proj_kernel<<<gA, 256, 0, stream>>>(key,   Wv, nullptr, vT,  nullptr, 2);

  kv_kernel<<<dim3(NB * NH), 256, 0, stream>>>(pkT, vT, kvT, ksum);

  attn_out_kernel<<<dim3(LSEQ / 128, NB * NH), 256, 0, stream>>>(pq, kvT, ksum, qmask, out);
}
